// RoGPELinearNodeEncoder_37684043055138
// MI455X (gfx1250) — compile-verified, matrix operand order checked by reading, not run on hardware
//
#include <hip/hip_runtime.h>
#include <hip/hip_bf16.h>


#define NN  8192
#define DD  128

#define F1  0.13533528323661270231f
#define F2  0.01831563888873418029f
#define F3  0.00247875217666635842f

typedef __attribute__((ext_vector_type(16))) __bf16 v16bf;
typedef __attribute__((ext_vector_type(8)))  __bf16 v8bf;
typedef __attribute__((ext_vector_type(8)))  float  v8f;
typedef __attribute__((ext_vector_type(8)))  int    v8i;
typedef __attribute__((ext_vector_type(2)))  int    v2i;

__device__ __forceinline__ v16bf load_frag_bf16(const __bf16* __restrict__ base,
                                                int ld, int lane) {
  int r    = lane & 15;
  int koff = (lane >> 4) << 3;
  const __bf16* p = base + (size_t)r * ld + koff;
  union { v16bf v; v8bf h[2]; } u;
  u.h[0] = *(const v8bf*)(p);
  u.h[1] = *(const v8bf*)(p + 16);
  return u.v;
}

__device__ __forceinline__ v8i load_frag_u8(const unsigned char* base,
                                            int ld, int lane) {
  int r    = lane & 15;
  int koff = (lane >> 4) << 3;
  const unsigned char* p = base + r * ld + koff;
  v2i c0 = *(const v2i*)(p +  0);
  v2i c1 = *(const v2i*)(p + 16);
  v2i c2 = *(const v2i*)(p + 32);
  v2i c3 = *(const v2i*)(p + 48);
  v8i f;
  f[0] = c0[0]; f[1] = c0[1]; f[2] = c1[0]; f[3] = c1[1];
  f[4] = c2[0]; f[5] = c2[1]; f[6] = c3[0]; f[7] = c3[1];
  return f;
}

__global__ __launch_bounds__(256)
void transpose_weights(const float* __restrict__ W0, const float* __restrict__ W1,
                       const float* __restrict__ W2, __bf16* __restrict__ Wt) {
  int id  = blockIdx.x * 256 + threadIdx.x;
  int w   = id >> 14;
  int rem = id & 16383;
  int n   = rem >> 7;
  int k   = rem & 127;
  const float* W = (w == 0) ? W0 : ((w == 1) ? W1 : W2);
  Wt[id] = (__bf16)W[k * DD + n];
}

__global__ __launch_bounds__(256)
void convert_bf16(const float* __restrict__ in, __bf16* __restrict__ out, int n) {
  int i = blockIdx.x * 256 + threadIdx.x;
  if (i < n) out[i] = (__bf16)in[i];
}

__global__ __launch_bounds__(256)
void mlp_gemm_bf16(const __bf16* __restrict__ A, const __bf16* __restrict__ Wt,
                   __bf16* __restrict__ Out) {
  int wave  = threadIdx.x >> 5;
  int lane  = threadIdx.x & 31;
  int mbase = blockIdx.x * 128 + wave * 16;

  v8f acc[8] = {};
#pragma unroll
  for (int ks = 0; ks < DD; ks += 32) {
    v16bf a = load_frag_bf16(A + (size_t)mbase * DD + ks, DD, lane);
#pragma unroll
    for (int nt = 0; nt < 8; ++nt) {
      v16bf b = load_frag_bf16(Wt + (size_t)(nt * 16) * DD + ks, DD, lane);
      acc[nt] = __builtin_amdgcn_wmma_f32_16x16x32_bf16(
          false, a, false, b, (short)0, acc[nt], false, false);
    }
  }
  int mrow = mbase + ((lane >> 4) << 3);
  int ncol = lane & 15;
#pragma unroll
  for (int nt = 0; nt < 8; ++nt)
#pragma unroll
    for (int r = 0; r < 8; ++r) {
      float v = acc[nt][r];
      v = v > 0.0f ? v : 0.0f;
      Out[(size_t)(mrow + r) * DD + nt * 16 + ncol] = (__bf16)v;
    }
}

__global__ __launch_bounds__(256)
void final_dot(const __bf16* __restrict__ h3, const float* __restrict__ W3,
               float* __restrict__ angles) {
  int wave = threadIdx.x >> 5;
  int lane = threadIdx.x & 31;
  int row  = blockIdx.x * 8 + wave;
  float s = 0.0f;
#pragma unroll
  for (int k = lane; k < DD; k += 32)
    s += (float)h3[(size_t)row * DD + k] * W3[k];
  for (int off = 16; off > 0; off >>= 1) s += __shfl_down(s, off, 32);
  if (lane == 0) angles[row] = s;
}

__global__ __launch_bounds__(256)
void build_adj(const int* __restrict__ ei, const float* __restrict__ angles,
               unsigned char* __restrict__ adj, unsigned char* __restrict__ adjT,
               float* __restrict__ step, int E) {
  int e = blockIdx.x * 256 + threadIdx.x;
  if (e >= E) return;
  int r = ei[e];
  int c = ei[E + e];
  adj [(size_t)r * NN + c] = 1;
  adjT[(size_t)c * NN + r] = 1;
  atomicAdd(&step[r], angles[c] * F1);
}

__global__ __launch_bounds__(256)
void bool_gemm_iu8(const unsigned char* __restrict__ A,
                   const unsigned char* __restrict__ Bt,
                   unsigned char* __restrict__ Out) {
  __shared__ __align__(16) unsigned char As[128 * 64];
  __shared__ __align__(16) unsigned char Bs[128 * 64];

  int wave = threadIdx.x >> 5;
  int lane = threadIdx.x & 31;
  int wm   = wave >> 1;
  int wn   = wave & 1;
  size_t mblock = (size_t)blockIdx.y * 128;
  size_t nblock = (size_t)blockIdx.x * 128;

  int srow = wave * 16 + (lane >> 2);
  int scol = (lane & 3) * 16;
  unsigned int ldsA0 = (unsigned int)(uintptr_t)(As + srow * 64 + scol);
  unsigned int ldsA1 = ldsA0 + 8 * 64;
  unsigned int ldsB0 = (unsigned int)(uintptr_t)(Bs + srow * 64 + scol);
  unsigned int ldsB1 = ldsB0 + 8 * 64;
  const unsigned char* gA = A  + (mblock + srow) * (size_t)NN + scol;
  const unsigned char* gB = Bt + (nblock + srow) * (size_t)NN + scol;

  v8i acc[2][4] = {};

  for (int k0 = 0; k0 < NN; k0 += 64) {
    unsigned long long a0 = (unsigned long long)(uintptr_t)(gA + k0);
    unsigned long long a1 = a0 + 8ull * NN;
    unsigned long long b0 = (unsigned long long)(uintptr_t)(gB + k0);
    unsigned long long b1 = b0 + 8ull * NN;
    asm volatile(
        "global_load_async_to_lds_b128 %0, %4, off\n\t"
        "global_load_async_to_lds_b128 %1, %5, off\n\t"
        "global_load_async_to_lds_b128 %2, %6, off\n\t"
        "global_load_async_to_lds_b128 %3, %7, off"
        :: "v"(ldsA0), "v"(ldsA1), "v"(ldsB0), "v"(ldsB1),
           "v"(a0), "v"(a1), "v"(b0), "v"(b1)
        : "memory");
    if (k0 + 64 < NN) {
      __builtin_prefetch(gA + k0 + 64, 0, 1);
      __builtin_prefetch(gB + k0 + 64, 0, 1);
    }
    asm volatile("s_wait_asynccnt 0x0" ::: "memory");
    __syncthreads();

    v8i a0f = load_frag_u8(As + (wm * 32 +  0) * 64, 64, lane);
    v8i a1f = load_frag_u8(As + (wm * 32 + 16) * 64, 64, lane);
#pragma unroll
    for (int nt = 0; nt < 4; ++nt) {
      v8i b = load_frag_u8(Bs + (wn * 64 + nt * 16) * 64, 64, lane);
      acc[0][nt] = __builtin_amdgcn_wmma_i32_16x16x64_iu8(
          false, a0f, false, b, acc[0][nt], false, false);
      acc[1][nt] = __builtin_amdgcn_wmma_i32_16x16x64_iu8(
          false, a1f, false, b, acc[1][nt], false, false);
    }
    __syncthreads();
  }

  size_t mrow = mblock + wm * 32 + ((lane >> 4) << 3);
  size_t ncol = nblock + wn * 64 + (lane & 15);
#pragma unroll
  for (int mt = 0; mt < 2; ++mt)
#pragma unroll
    for (int nt = 0; nt < 4; ++nt)
#pragma unroll
      for (int r = 0; r < 8; ++r)
        Out[(mrow + mt * 16 + r) * (size_t)NN + ncol + nt * 16] =
            (unsigned char)(acc[mt][nt][r] > 0);
}

__global__ __launch_bounds__(256)
void spmv_u8(const unsigned char* __restrict__ P, const float* __restrict__ x,
             float* __restrict__ y) {
  int wave = threadIdx.x >> 5;
  int lane = threadIdx.x & 31;
  int row  = blockIdx.x * 8 + wave;
  const unsigned char* rp = P + (size_t)row * NN;
  float s = 0.0f;
  for (int j0 = 0; j0 < NN; j0 += 128) {
    int j = j0 + lane * 4;
    unsigned int m = *(const unsigned int*)(rp + j);
    if (m) {
      if (m & 0x000000ffu) s += x[j + 0];
      if (m & 0x0000ff00u) s += x[j + 1];
      if (m & 0x00ff0000u) s += x[j + 2];
      if (m & 0xff000000u) s += x[j + 3];
    }
  }
  for (int off = 16; off > 0; off >>= 1) s += __shfl_down(s, off, 32);
  if (lane == 0) y[row] = s;
}

__global__ __launch_bounds__(256)
void vec_add(const float* __restrict__ a, const float* __restrict__ b,
             float* __restrict__ out) {
  int i = blockIdx.x * 256 + threadIdx.x;
  if (i < NN) out[i] = a[i] + b[i];
}

__global__ __launch_bounds__(256)
void vec_update(const float* __restrict__ tmp, float* __restrict__ step,
                const float* __restrict__ enh_in, float* __restrict__ enh_out,
                float f) {
  int i = blockIdx.x * 256 + threadIdx.x;
  if (i < NN) {
    float s = step[i] + f * tmp[i];
    step[i]    = s;
    enh_out[i] = enh_in[i] + s;
  }
}

extern "C" void kernel_launch(void* const* d_in, const int* in_sizes, int n_in,
                              void* d_out, int out_size, void* d_ws, size_t ws_size,
                              hipStream_t stream) {
  const float* coeffs = (const float*)d_in[0];
  const float* W0     = (const float*)d_in[1];
  const float* W1     = (const float*)d_in[2];
  const float* W2     = (const float*)d_in[3];
  const float* W3     = (const float*)d_in[4];
  const int*   eidx   = (const int*)d_in[5];
  const int    E      = in_sizes[5] / 2;

  unsigned char* ws   = (unsigned char*)d_ws;
  const size_t adjB   = (size_t)NN * NN;
  unsigned char* adj  = ws;
  unsigned char* adjT = ws + adjB;
  unsigned char* P2   = ws + 2 * adjB;
  __bf16* Wt = (__bf16*)(ws + 3 * adjB);
  __bf16* h0 = Wt + 3 * DD * DD;
  __bf16* h1 = h0 + (size_t)NN * DD;
  __bf16* h2 = h1 + (size_t)NN * DD;
  __bf16* h3 = h2 + (size_t)NN * DD;
  float* angles = (float*)(h3 + (size_t)NN * DD);
  float* step   = angles + NN;
  float* enh    = step + NN;
  float* tmp    = enh + NN;

  hipMemsetAsync(adj, 0, 2 * adjB, stream);
  hipMemsetAsync(step, 0, sizeof(float) * NN, stream);

  transpose_weights<<<(3 * DD * DD) / 256, 256, 0, stream>>>(W0, W1, W2, Wt);
  convert_bf16<<<(NN * DD) / 256, 256, 0, stream>>>(coeffs, h0, NN * DD);
  mlp_gemm_bf16<<<NN / 128, 256, 0, stream>>>(h0, Wt,               h1);
  mlp_gemm_bf16<<<NN / 128, 256, 0, stream>>>(h1, Wt + DD * DD,     h2);
  mlp_gemm_bf16<<<NN / 128, 256, 0, stream>>>(h2, Wt + 2 * DD * DD, h3);
  final_dot<<<NN / 8, 256, 0, stream>>>(h3, W3, angles);

  build_adj<<<(E + 255) / 256, 256, 0, stream>>>(eidx, angles, adj, adjT, step, E);
  vec_add<<<NN / 256, 256, 0, stream>>>(angles, step, enh);

  dim3 gg(NN / 128, NN / 128);
  bool_gemm_iu8<<<gg, 256, 0, stream>>>(adj, adjT, P2);
  spmv_u8<<<NN / 8, 256, 0, stream>>>(P2, enh, tmp);
  vec_update<<<NN / 256, 256, 0, stream>>>(tmp, step, enh, enh, F2);

  bool_gemm_iu8<<<gg, 256, 0, stream>>>(P2, adjT, adj);
  spmv_u8<<<NN / 8, 256, 0, stream>>>(adj, enh, tmp);
  vec_update<<<NN / 256, 256, 0, stream>>>(tmp, step, enh, (float*)d_out, F3);
}
